// Attention_91087666413717
// MI455X (gfx1250) — matrix operand order repaired and checked by reading, not compiled, not run on hardware
//
#include <hip/hip_runtime.h>
#include <math.h>

typedef _Float16 h16 __attribute__((ext_vector_type(16)));
typedef _Float16 h4  __attribute__((ext_vector_type(4)));
typedef float    f8  __attribute__((ext_vector_type(8)));
typedef int      v4i __attribute__((ext_vector_type(4)));

#if __has_builtin(__builtin_amdgcn_global_load_async_to_lds_b128)
#define HAVE_ASYNC_LDS 1
#else
#define HAVE_ASYNC_LDS 0
#endif

__device__ __forceinline__ void wait_async0() {
#if __has_builtin(__builtin_amdgcn_s_wait_asynccnt)
  __builtin_amdgcn_s_wait_asynccnt(0);
#else
  asm volatile("s_wait_asynccnt 0" ::: "memory");
#endif
}

__device__ __forceinline__ f8 wmma32(h16 a, h16 b, f8 c) {
  // D = A(16x32 f16) * B(32x16 f16) + C(16x16 f32)
  return __builtin_amdgcn_wmma_f32_16x16x32_f16(false, a, false, b, (short)0, c,
                                                false, false);
}

union FragU { uint4 q[2]; h16 h; };

// A-fragment (16x32 f16): lane row = lane&15; kg = lane>>4;
// halves 0-7 -> K = 8*kg.. ; halves 8-15 -> K = 16+8*kg..   (two contiguous 16B chunks)
__device__ __forceinline__ h16 load_frag_a(const _Float16* p, int row0, int ld, int lane) {
  const int row = row0 + (lane & 15);
  const int kg  = lane >> 4;
  const char* base = (const char*)(p + (size_t)row * ld);
  FragU u;
  u.q[0] = *(const uint4*)(base + (size_t)(8 * kg) * 2);
  u.q[1] = *(const uint4*)(base + (size_t)(16 + 8 * kg) * 2);
  return u.h;
}

// B-fragment (32x16 f16): lane col = lane&15; halves 0-7 -> K = 8*kg..; halves 8-15 -> K = 16+8*kg..
// p points at a column-major-in-K tile ([N][K] contiguous in K); ld = K stride.
__device__ __forceinline__ h16 load_frag_b(const _Float16* p, int col0, int ld, int lane) {
  const int col = col0 + (lane & 15);
  const int kg  = lane >> 4;
  const char* base = (const char*)(p + (size_t)col * ld);
  FragU u;
  u.q[0] = *(const uint4*)(base + (size_t)(8 * kg) * 2);
  u.q[1] = *(const uint4*)(base + (size_t)(16 + 8 * kg) * 2);
  return u.h;
}

// ---------------- fp32 -> fp16 convert (x4 vectorized) ----------------
__global__ __launch_bounds__(256) void cvt4_kernel(const float* __restrict__ in,
                                                   _Float16* __restrict__ out, int n4) {
  int i = blockIdx.x * 256 + threadIdx.x;
  if (i < n4) {
    float4 v = ((const float4*)in)[i];
    h4 o; o.x = (_Float16)v.x; o.y = (_Float16)v.y; o.z = (_Float16)v.z; o.w = (_Float16)v.w;
    ((h4*)out)[i] = o;
  }
}

// ---------------- tiled WMMA GEMM: C[M,N] = A[M,K] * B[K,N] ----------------
// Block tile 64(M) x 256(N), 8 waves (2x4), wave tile 32x64 (2x4 WMMA tiles = 8 wmma/step),
// KB=32, double-buffered LDS with register prefetch (one barrier per K-step).
// MODE 0: scatter into qkv layout [3][B][H][N][64] f16, scaling Q by 1/8.
// MODE 1: f32 + bias.
template <int MODE>
__global__ __launch_bounds__(256) void gemm_wmma_kernel(
    const _Float16* __restrict__ A, const _Float16* __restrict__ B,
    _Float16* __restrict__ outH, float* __restrict__ outF,
    const float* __restrict__ bias, int M, int N, int K) {
  __shared__ __align__(16) _Float16 Asm[2][64 * 32];    // [m][k] row-major
  __shared__ __align__(16) _Float16 Bsm[2][256 * 32];   // transposed: [n][k]
  const int tid  = threadIdx.x;
  const int lane = tid & 31;
  const int w    = tid >> 5;
  const int wr   = w >> 2;   // 0..1
  const int wc   = w & 3;    // 0..3
  const int m0   = blockIdx.x * 64;
  const int n0   = blockIdx.y * 256;
  const int aRow = tid >> 2, aC4 = tid & 3;

  f8 acc[2][4] = {};
  uint4 aReg;
  uint4 bReg[4];

  auto fetch = [&](int k0) {
    aReg = *((const uint4*)(A + (size_t)(m0 + aRow) * K + k0) + aC4);
    #pragma unroll
    for (int i = 0; i < 4; ++i) {
      int idx = tid + i * 256;            // 0..1023
      int kk = idx >> 5, nc = idx & 31;   // kk 0..31, nc covers 8 cols
      bReg[i] = *(const uint4*)(B + (size_t)(k0 + kk) * N + n0 + nc * 8);
    }
  };
  auto stage = [&](int buf) {
    *((uint4*)(Asm[buf] + aRow * 32) + aC4) = aReg;
    #pragma unroll
    for (int i = 0; i < 4; ++i) {
      int idx = tid + i * 256;
      int kk = idx >> 5, nc = idx & 31;
      const _Float16* hp = (const _Float16*)&bReg[i];
      #pragma unroll
      for (int j = 0; j < 8; ++j) Bsm[buf][(nc * 8 + j) * 32 + kk] = hp[j];
    }
  };

  const int steps = K / 32;
  fetch(0);
  stage(0);
  __syncthreads();

  for (int s = 0; s < steps; ++s) {
    const int cur = s & 1;
    if (s + 1 < steps) fetch((s + 1) * 32);   // prefetch next tile into registers

    h16 a0 = load_frag_a(Asm[cur], wr * 32 + 0,  32, lane);
    h16 a1 = load_frag_a(Asm[cur], wr * 32 + 16, 32, lane);
    #pragma unroll
    for (int tc = 0; tc < 4; ++tc) {
      h16 bf = load_frag_b(Bsm[cur], wc * 64 + tc * 16, 32, lane);
      acc[0][tc] = wmma32(a0, bf, acc[0][tc]);
      acc[1][tc] = wmma32(a1, bf, acc[1][tc]);
    }
    if (s + 1 < steps) {
      stage(cur ^ 1);
      __syncthreads();
    }
  }

  // epilogue: C frag -> lane holds row r+8*(lane>>4), col lane&15
  const int g = lane >> 4, c = lane & 15;
  #pragma unroll
  for (int tr = 0; tr < 2; ++tr)
    #pragma unroll
    for (int tc = 0; tc < 4; ++tc)
      #pragma unroll
      for (int r = 0; r < 8; ++r) {
        int gm = m0 + wr * 32 + tr * 16 + r + 8 * g;
        int ge = n0 + wc * 64 + tc * 16 + c;
        float v = acc[tr][tc][r];
        if (MODE == 0) {
          int t = ge / 768, rem = ge % 768;
          int hh = rem >> 6, dd = rem & 63;
          int b = gm >> 10, n = gm & 1023;
          if (t == 0) v *= 0.125f;  // fold 1/sqrt(64) into Q
          size_t dst = ((((size_t)t * 16 + b) * 12 + hh) * 1024 + n) * 64 + dd;
          outH[dst] = (_Float16)v;
        } else {
          outF[(size_t)gm * N + ge] = v + bias[ge];
        }
      }
}

// ---------------- fused flash attention ----------------
// Grid (qtile=16, h=12, b=16); block 128 = 4 waves; wave owns 16 query rows.
// 64-key tiles: 16 WMMAs per iteration per wave; K staged async-to-LDS (when available),
// V transposed at staging so P@V B-fragments are contiguous ds_load_b128.
__global__ __launch_bounds__(128) void flash_kernel(
    const _Float16* __restrict__ qh, const _Float16* __restrict__ kh,
    const _Float16* __restrict__ vh, _Float16* __restrict__ outh) {
  __shared__ __align__(16) _Float16 Klds[64 * 64];     // [key][d]
  __shared__ __align__(16) _Float16 Vt[64 * 64];       // [d][key]
  __shared__ __align__(16) _Float16 Plds[4][16 * 64];
  const int tid  = threadIdx.x;
  const int lane = tid & 31;
  const int w    = tid >> 5;
  const int h    = blockIdx.y;
  const int b    = blockIdx.z;
  const size_t bh = ((size_t)b * 12 + h) * 1024 * 64;
  const int q0 = blockIdx.x * 64 + w * 16;

  // Q fragments for this wave's 16 rows (hd=64 -> two K=32 A-frags), from global
  h16 aq0 = load_frag_a(qh + bh + (size_t)q0 * 64, 0, 64, lane);
  h16 aq1 = load_frag_a(qh + bh + (size_t)q0 * 64 + 32, 0, 64, lane);

  f8 accO[4] = {};
  float m[8], l[8];
  #pragma unroll
  for (int r = 0; r < 8; ++r) { m[r] = -1e30f; l[r] = 0.0f; }

  for (int kt = 0; kt < 1024; kt += 64) {
    __syncthreads();
    // cooperative stage: K row-major (async path), V transposed (manual)
    #pragma unroll
    for (int i = 0; i < 4; ++i) {
      int idx = tid + i * 128;   // 0..511 : 64 rows x 8 16B-chunks
      int row = idx >> 3, c8 = idx & 7;
      const _Float16* gk = kh + bh + (size_t)(kt + row) * 64 + c8 * 8;
      _Float16* lk = Klds + row * 64 + c8 * 8;
#if HAVE_ASYNC_LDS
      __builtin_amdgcn_global_load_async_to_lds_b128((v4i*)gk, (v4i*)lk, 0, 0);
#else
      *(uint4*)lk = *(const uint4*)gk;
#endif
      uint4 d = *((const uint4*)(vh + bh + (size_t)(kt + row) * 64) + c8);
      const _Float16* hp = (const _Float16*)&d;
      #pragma unroll
      for (int j = 0; j < 8; ++j) Vt[(c8 * 8 + j) * 64 + row] = hp[j];
    }
#if HAVE_ASYNC_LDS
    wait_async0();
#endif
    __syncthreads();

    // S = Q * K^T : four 16-key column groups, K-dim 64 = 2 chained wmma each
    f8 z = {};
    f8 sArr[4];
    #pragma unroll
    for (int j = 0; j < 4; ++j) {
      sArr[j] = wmma32(aq0, load_frag_b(Klds,      j * 16, 64, lane), z);
      sArr[j] = wmma32(aq1, load_frag_b(Klds + 32, j * 16, 64, lane), sArr[j]);
    }

    // online softmax across 64 keys
    float rmax[8];
    #pragma unroll
    for (int r = 0; r < 8; ++r) {
      float v = fmaxf(fmaxf(sArr[0][r], sArr[1][r]), fmaxf(sArr[2][r], sArr[3][r]));
      #pragma unroll
      for (int msk = 1; msk < 16; msk <<= 1) v = fmaxf(v, __shfl_xor(v, msk, 32));
      rmax[r] = v;
    }
    const int g = lane >> 4, c = lane & 15;
    #pragma unroll
    for (int r = 0; r < 8; ++r) {
      float mn    = fmaxf(m[r], rmax[r]);
      float alpha = __expf(m[r] - mn);
      m[r] = mn;
      float pj[4], ps = 0.0f;
      #pragma unroll
      for (int j = 0; j < 4; ++j) { pj[j] = __expf(sArr[j][r] - mn); ps += pj[j]; }
      #pragma unroll
      for (int msk = 1; msk < 16; msk <<= 1) ps += __shfl_xor(ps, msk, 32);
      l[r] = l[r] * alpha + ps;
      #pragma unroll
      for (int t = 0; t < 4; ++t) accO[t][r] *= alpha;
      int row = r + 8 * g;
      #pragma unroll
      for (int j = 0; j < 4; ++j) Plds[w][row * 64 + j * 16 + c] = (_Float16)pj[j];
    }
    asm volatile("s_wait_dscnt 0" ::: "memory");  // per-wave LDS RAW fence for P restage

    // O += P(16x64) * V(64x64): two K=32 chunks x four d-groups
    h16 ap0 = load_frag_a(&Plds[w][0],      0, 64, lane);
    h16 ap1 = load_frag_a(&Plds[w][0] + 32, 0, 64, lane);
    #pragma unroll
    for (int t = 0; t < 4; ++t) {
      accO[t] = wmma32(ap0, load_frag_b(Vt,      t * 16, 64, lane), accO[t]);
      accO[t] = wmma32(ap1, load_frag_b(Vt + 32, t * 16, 64, lane), accO[t]);
    }
  }

  // normalize and write to [B, N, H*64] f16 (input of the out-projection GEMM)
  const int g = lane >> 4, c = lane & 15;
  #pragma unroll
  for (int t = 0; t < 4; ++t)
    #pragma unroll
    for (int r = 0; r < 8; ++r) {
      int row = q0 + r + 8 * g;
      int col = h * 64 + t * 16 + c;
      outh[((size_t)b * 1024 + row) * 768 + col] = (_Float16)(accO[t][r] / l[r]);
    }
}

extern "C" void kernel_launch(void* const* d_in, const int* in_sizes, int n_in,
                              void* d_out, int out_size, void* d_ws, size_t ws_size,
                              hipStream_t stream) {
  const float* x      = (const float*)d_in[0];
  const float* qkv_w  = (const float*)d_in[1];
  const float* proj_w = (const float*)d_in[2];
  const float* proj_b = (const float*)d_in[3];
  float* out = (float*)d_out;

  const int B = 16, N = 1024, D = 768;
  const size_t xN    = (size_t)B * N * D;          // 12,582,912
  const size_t wqkvN = (size_t)D * 3 * D;          // 1,769,472
  const size_t wprjN = (size_t)D * D;              // 589,824
  const size_t BHND  = (size_t)B * 12 * N * 64;    // 12,582,912

  char* ws = (char*)d_ws;
  _Float16* xh    = (_Float16*)ws;                 ws += xN * 2;
  _Float16* wqkvh = (_Float16*)ws;                 ws += wqkvN * 2;
  _Float16* wprjh = (_Float16*)ws;                 ws += wprjN * 2;
  _Float16* qkvh  = (_Float16*)ws;                 ws += 3 * BHND * 2;
  _Float16* attnh = (_Float16*)ws;                 ws += xN * 2;

  // fp32 -> fp16
  cvt4_kernel<<<(int)(xN / 4 / 256), 256, 0, stream>>>(x, xh, (int)(xN / 4));
  cvt4_kernel<<<(int)(wqkvN / 4 / 256), 256, 0, stream>>>(qkv_w, wqkvh, (int)(wqkvN / 4));
  cvt4_kernel<<<(int)(wprjN / 4 / 256), 256, 0, stream>>>(proj_w, wprjh, (int)(wprjN / 4));

  // QKV projection: [16384,768] x [768,2304] -> q/k/v [B,H,N,64] f16 (Q pre-scaled)
  gemm_wmma_kernel<0><<<dim3(256, 9), 256, 0, stream>>>(
      xh, wqkvh, qkvh, nullptr, nullptr, B * N, 3 * D, D);

  // fused flash attention
  flash_kernel<<<dim3(16, 12, 16), 128, 0, stream>>>(
      qkvh, qkvh + BHND, qkvh + 2 * BHND, attnh);

  // output projection: [16384,768] x [768,768] + bias -> f32 out
  gemm_wmma_kernel<1><<<dim3(256, 3), 256, 0, stream>>>(
      attnh, wprjh, nullptr, out, proj_b, B * N, D, D);
}
